// MixtralSparseMoeBlock_78331613545178
// MI455X (gfx1250) — compile-verified, matrix operand order checked by reading, not run on hardware
//
#include <hip/hip_runtime.h>
#include <math.h>

// ---------------------------------------------------------------------------
// Mixtral sparse MoE block for MI455X (gfx1250, wave32, WMMA, TDM).
// Sparse top-2 routing -> per-expert token gather -> fused SwiGLU GEMM (bf16
// WMMA, f32 accum, double-buffered LDS) -> down-proj GEMM (TDM async A-tile)
// -> weighted atomic combine.
// ---------------------------------------------------------------------------

#define T_TOK 2048
#define H_DIM 2048
#define F_DIM 1408
#define E_NUM 8

typedef __attribute__((ext_vector_type(16))) __bf16          bf16x16;
typedef __attribute__((ext_vector_type(8)))  __bf16          bf16x8;
typedef __attribute__((ext_vector_type(8)))  float           f32x8;
typedef __attribute__((ext_vector_type(8)))  unsigned short  u16x8;
typedef __attribute__((ext_vector_type(4)))  unsigned short  u16x4;
typedef __attribute__((ext_vector_type(4)))  unsigned int    u32x4;
typedef __attribute__((ext_vector_type(8)))  int             i32x8;
typedef __attribute__((ext_vector_type(4)))  int             i32x4;

#if __has_builtin(__builtin_amdgcn_tensor_load_to_lds) && \
    __has_builtin(__builtin_amdgcn_s_wait_tensorcnt)
#define HAVE_TDM 1
#else
#define HAVE_TDM 0
#endif

__device__ __forceinline__ unsigned short f2bf(float f) {
    // native fp32 -> bf16 (round-to-nearest-even), single v_cvt op
    __bf16 b = (__bf16)f;
    return __builtin_bit_cast(unsigned short, b);
}

__device__ __forceinline__ f32x8 zero8() {
    f32x8 z;
#pragma unroll
    for (int i = 0; i < 8; ++i) z[i] = 0.f;
    return z;
}

__device__ __forceinline__ f32x8 wmma_bf16(bf16x16 a, bf16x16 b, f32x8 c) {
    return __builtin_amdgcn_wmma_f32_16x16x32_bf16(
        false, a, false, b, (short)0, c, false, false);
}

// Assemble a 16x32 bf16 fragment from an LDS tile stored [row][32] (row = M
// for A, N for B).  ISA layout: lanes 0-15 take K[0..8)+K[16..24); lanes
// 16-31 take K[8..16)+K[24..32).  All chunks 16B aligned -> ds_load_b128.
__device__ __forceinline__ bf16x16 load_frag(const unsigned short* rowp, int kgrp) {
    u16x8 c0 = *(const u16x8*)(rowp + kgrp * 8);
    u16x8 c1 = *(const u16x8*)(rowp + 16 + kgrp * 8);
    bf16x8 b0 = __builtin_bit_cast(bf16x8, c0);
    bf16x8 b1 = __builtin_bit_cast(bf16x8, c1);
    return __builtin_shufflevector(b0, b1, 0, 1, 2, 3, 4, 5, 6, 7,
                                   8, 9, 10, 11, 12, 13, 14, 15);
}

#if HAVE_TDM
// LDS byte offset of a __shared__ object (generic -> AS(3) -> 32-bit offset).
__device__ __forceinline__ unsigned lds_addr_of(const void* p) {
    return (unsigned)(unsigned long long)
        (__attribute__((address_space(3))) const void*)p;
}

// Issue a 2D TDM load: tile_h rows x tile_w bf16 elements, row stride
// `stride` elements, into LDS at lds_addr.  D# layout per CDNA5 ISA ch.8.
// This toolchain's builtin takes 6 args: (g0, g1, g2, g3, g4, cpol).
__device__ __forceinline__ void tdm_load_2d(unsigned lds_addr, const void* gptr,
                                            unsigned tile_w, unsigned tile_h,
                                            unsigned long long stride) {
    unsigned long long ga = (unsigned long long)gptr;
    u32x4 g0;
    g0[0] = 1u;                                      // count=1 (valid), user mode
    g0[1] = lds_addr;                                // lds_addr [63:32]
    g0[2] = (unsigned)(ga & 0xFFFFFFFFull);          // global_addr [95:64]
    g0[3] = (unsigned)((ga >> 32) & 0x1FFFFFFull)    // global_addr [120:96]
          | (2u << 30);                              // type=2 ("image") [127:126]
    i32x8 g1;
    g1[0] = (int)(1u << 16);                         // data_size=1 (2 bytes)
    g1[1] = (int)((tile_w & 0xFFFFu) << 16);         // tensor_dim0 lo
    g1[2] = (int)((tile_w >> 16) | ((tile_h & 0xFFFFu) << 16)); // d0 hi | d1 lo
    g1[3] = (int)((tile_h >> 16) | ((tile_w & 0xFFFFu) << 16)); // d1 hi | tile_dim0
    g1[4] = (int)(tile_h & 0xFFFFu);                 // tile_dim1 (tile_dim2=0)
    g1[5] = (int)(unsigned)(stride & 0xFFFFFFFFull); // tensor_dim0_stride lo
    g1[6] = (int)(unsigned)((stride >> 32) & 0xFFFFull);
    g1[7] = 0;
    i32x4 gz4;
    gz4[0] = 0; gz4[1] = 0; gz4[2] = 0; gz4[3] = 0;
    i32x8 gz8;
#pragma unroll
    for (int i = 0; i < 8; ++i) gz8[i] = 0;
    __builtin_amdgcn_tensor_load_to_lds(g0, g1, gz4, gz4, gz8, 0);
}
#endif

// ---------------------------------------------------------------------------
// Kernel 1: zero output hidden-state region and expert counters.
// ---------------------------------------------------------------------------
__global__ __launch_bounds__(256) void zero_init(float* __restrict__ out,
                                                 int n4, int* __restrict__ cnt) {
    int i = blockIdx.x * 256 + threadIdx.x;
    if (i < n4) {
        float4 z = {0.f, 0.f, 0.f, 0.f};
        *(float4*)(out + (size_t)i * 4) = z;
    }
    if (i < E_NUM) cnt[i] = 0;
}

// ---------------------------------------------------------------------------
// Kernel 2: router.  One wave32 per token.
// ---------------------------------------------------------------------------
__global__ __launch_bounds__(256) void router_kernel(
    const float* __restrict__ x, const float* __restrict__ gw,
    float* __restrict__ logits_out, int* __restrict__ cnt,
    int* __restrict__ tok_idx, float* __restrict__ tok_w) {
    const int lane = threadIdx.x & 31;
    const int wid  = threadIdx.x >> 5;
    const int t    = blockIdx.x * 8 + wid;

    float acc[E_NUM];
#pragma unroll
    for (int e = 0; e < E_NUM; ++e) acc[e] = 0.f;

    const float* xr = x + (size_t)t * H_DIM;
    for (int i = lane; i < H_DIM; i += 32) {
        float xv = xr[i];
#pragma unroll
        for (int e = 0; e < E_NUM; ++e) acc[e] += xv * gw[e * H_DIM + i];
    }
#pragma unroll
    for (int e = 0; e < E_NUM; ++e)
        for (int off = 16; off; off >>= 1) acc[e] += __shfl_xor(acc[e], off, 32);

    if (lane == 0) {
        float m = acc[0];
#pragma unroll
        for (int e = 1; e < E_NUM; ++e) m = fmaxf(m, acc[e]);
        float ex[E_NUM], s = 0.f;
#pragma unroll
        for (int e = 0; e < E_NUM; ++e) { ex[e] = __expf(acc[e] - m); s += ex[e]; }
        float inv = 1.f / s;

        int i0 = 0;
#pragma unroll
        for (int e = 1; e < E_NUM; ++e) if (ex[e] > ex[i0]) i0 = e;
        int i1 = (i0 == 0) ? 1 : 0;
#pragma unroll
        for (int e = 0; e < E_NUM; ++e)
            if (e != i0 && ex[e] > ex[i1]) i1 = e;

        float p0 = ex[i0] * inv, p1 = ex[i1] * inv;
        float rs = 1.f / (p0 + p1);

#pragma unroll
        for (int e = 0; e < E_NUM; ++e) logits_out[t * E_NUM + e] = acc[e];

        int pos0 = atomicAdd(&cnt[i0], 1);
        tok_idx[i0 * T_TOK + pos0] = t;
        tok_w[i0 * T_TOK + pos0]   = p0 * rs;
        int pos1 = atomicAdd(&cnt[i1], 1);
        tok_idx[i1 * T_TOK + pos1] = t;
        tok_w[i1 * T_TOK + pos1]   = p1 * rs;
    }
}

// ---------------------------------------------------------------------------
// Kernel 3: x f32 -> bf16 once.
// ---------------------------------------------------------------------------
__global__ __launch_bounds__(256) void convert_x(const float* __restrict__ x,
                                                 unsigned short* __restrict__ xb) {
    size_t i = ((size_t)blockIdx.x * 256 + threadIdx.x) * 4;
    float4 v = *(const float4*)(x + i);
    u16x4 b = {f2bf(v.x), f2bf(v.y), f2bf(v.z), f2bf(v.w)};
    *(u16x4*)(xb + i) = b;
}

// ---------------------------------------------------------------------------
// Kernel 4: prefix-sum over 8 expert counts.
// ---------------------------------------------------------------------------
__global__ void scan_base(const int* __restrict__ cnt, int* __restrict__ base) {
    if (threadIdx.x == 0 && blockIdx.x == 0) {
        int s = 0;
        for (int e = 0; e < E_NUM; ++e) { base[e] = s; s += cnt[e]; }
    }
}

// ---------------------------------------------------------------------------
// Kernel 5: fused gate/up GEMM + SwiGLU.  C tile 128(tok) x 64(F), 8 waves
// of 32x32, dual accumulators, double-buffered LDS, hoisted gather addresses.
// ---------------------------------------------------------------------------
__global__ __launch_bounds__(256) void moe_gemm1(
    const unsigned short* __restrict__ xb, const int* __restrict__ tok_idx,
    const int* __restrict__ cnt_arr, const int* __restrict__ base_arr,
    const float* __restrict__ w1, const float* __restrict__ w3,
    unsigned short* __restrict__ hb) {
    __shared__ __attribute__((aligned(16))) unsigned short lA[2][128 * 32];
    __shared__ __attribute__((aligned(16))) unsigned short lB1[2][64 * 32];
    __shared__ __attribute__((aligned(16))) unsigned short lB3[2][64 * 32];

    const int e  = blockIdx.z;
    const int mt = blockIdx.y;
    const int nt = blockIdx.x;
    const int cnt = cnt_arr[e];
    if (mt * 128 >= cnt) return;           // uniform early-exit
    const int baseS = base_arr[e];

    const int tid  = threadIdx.x;
    const int lane = tid & 31;
    const int wid  = tid >> 5;
    const int wm   = (wid & 3) * 32;
    const int wn   = (wid >> 2) * 32;
    const int kgrp = lane >> 4;
    const int lrow = lane & 15;

    const int* tIdx = tok_idx + e * T_TOK;

    // Hoisted per-thread A-gather addresses (2 chunks of 16B per K-step).
    const int segA = tid & 3;
    int rowA[2];
    const unsigned short* aSrc[2];
#pragma unroll
    for (int j = 0; j < 2; ++j) {
        rowA[j] = j * 64 + (tid >> 2);
        int p = mt * 128 + rowA[j];
        int pc = (p < cnt) ? p : (cnt - 1);  // clamp: garbage rows discarded later
        int t = tIdx[pc];
        aSrc[j] = xb + (size_t)t * H_DIM + segA * 8;
    }
    // Hoisted B pointers (2 chunks of 4 f32 per matrix per K-step).
    const int segB = tid & 7;
    const float* b1Src[2];
    const float* b3Src[2];
#pragma unroll
    for (int j = 0; j < 2; ++j) {
        int row = j * 32 + (tid >> 3);
        size_t off = (size_t)e * F_DIM * H_DIM + (size_t)(nt * 64 + row) * H_DIM
                   + segB * 4;
        b1Src[j] = w1 + off;
        b3Src[j] = w3 + off;
    }

    f32x8 accG[2][2], accU[2][2];
#pragma unroll
    for (int i = 0; i < 2; ++i)
#pragma unroll
        for (int j = 0; j < 2; ++j) { accG[i][j] = zero8(); accU[i][j] = zero8(); }

    auto load_tile = [&](int buf, int k0) {
#pragma unroll
        for (int j = 0; j < 2; ++j) {
            u16x8 val = *(const u16x8*)(aSrc[j] + k0);
            *(u16x8*)(&lA[buf][rowA[j] * 32 + segA * 8]) = val;
        }
#pragma unroll
        for (int j = 0; j < 2; ++j) {
            int row = j * 32 + (tid >> 3);
            __builtin_prefetch(b1Src[j] + k0 + 96, 0, 1);   // ~3 K-tiles ahead
            __builtin_prefetch(b3Src[j] + k0 + 96, 0, 1);
            float4 v1 = *(const float4*)(b1Src[j] + k0);
            float4 v3 = *(const float4*)(b3Src[j] + k0);
            u16x4 b1v = {f2bf(v1.x), f2bf(v1.y), f2bf(v1.z), f2bf(v1.w)};
            u16x4 b3v = {f2bf(v3.x), f2bf(v3.y), f2bf(v3.z), f2bf(v3.w)};
            *(u16x4*)(&lB1[buf][row * 32 + segB * 4]) = b1v;
            *(u16x4*)(&lB3[buf][row * 32 + segB * 4]) = b3v;
        }
    };

    load_tile(0, 0);
    int cur = 0;
    for (int k0 = 0; k0 < H_DIM; k0 += 32) {
        __syncthreads();                       // tile `cur` is ready
        if (k0 + 32 < H_DIM) load_tile(cur ^ 1, k0 + 32);

        bf16x16 a0  = load_frag(&lA[cur][(wm + 0 * 16 + lrow) * 32], kgrp);
        bf16x16 a1  = load_frag(&lA[cur][(wm + 1 * 16 + lrow) * 32], kgrp);
        bf16x16 bg0 = load_frag(&lB1[cur][(wn + 0 * 16 + lrow) * 32], kgrp);
        bf16x16 bg1 = load_frag(&lB1[cur][(wn + 1 * 16 + lrow) * 32], kgrp);
        bf16x16 bu0 = load_frag(&lB3[cur][(wn + 0 * 16 + lrow) * 32], kgrp);
        bf16x16 bu1 = load_frag(&lB3[cur][(wn + 1 * 16 + lrow) * 32], kgrp);

        accG[0][0] = wmma_bf16(a0, bg0, accG[0][0]);
        accG[0][1] = wmma_bf16(a0, bg1, accG[0][1]);
        accG[1][0] = wmma_bf16(a1, bg0, accG[1][0]);
        accG[1][1] = wmma_bf16(a1, bg1, accG[1][1]);
        accU[0][0] = wmma_bf16(a0, bu0, accU[0][0]);
        accU[0][1] = wmma_bf16(a0, bu1, accU[0][1]);
        accU[1][0] = wmma_bf16(a1, bu0, accU[1][0]);
        accU[1][1] = wmma_bf16(a1, bu1, accU[1][1]);
        cur ^= 1;
    }

    // Epilogue: h = silu(g) * u, compacted bf16 store.
#pragma unroll
    for (int mi = 0; mi < 2; ++mi)
#pragma unroll
        for (int ni = 0; ni < 2; ++ni)
#pragma unroll
            for (int v = 0; v < 8; ++v) {
                int rowt = wm + mi * 16 + v + kgrp * 8;
                int p = mt * 128 + rowt;
                if (p < cnt) {
                    float g = accG[mi][ni][v];
                    float u = accU[mi][ni][v];
                    float hv = (g / (1.f + __expf(-g))) * u;
                    int col = nt * 64 + wn + ni * 16 + lrow;
                    hb[(size_t)(baseS + p) * F_DIM + col] = f2bf(hv);
                }
            }
}

// ---------------------------------------------------------------------------
// Kernel 6: down-projection h @ w2^T.  A tile is contiguous compacted rows
// -> loaded by the Tensor Data Mover (one 2D DMA per K-step, wave 0 issues,
// s_wait_tensorcnt completes).  Epilogue: routing_weight * acc atomic scatter.
// ---------------------------------------------------------------------------
__global__ __launch_bounds__(256) void moe_gemm2(
    const unsigned short* __restrict__ hb, const int* __restrict__ tok_idx,
    const float* __restrict__ tok_w, const int* __restrict__ cnt_arr,
    const int* __restrict__ base_arr, const float* __restrict__ w2,
    float* __restrict__ out) {
    __shared__ __attribute__((aligned(16))) unsigned short lA[2][128 * 32];
    __shared__ __attribute__((aligned(16))) unsigned short lB[2][64 * 32];

    const int e  = blockIdx.z;
    const int mt = blockIdx.y;
    const int nt = blockIdx.x;
    const int cnt = cnt_arr[e];
    if (mt * 128 >= cnt) return;
    const int baseS = base_arr[e];

    const int tid  = threadIdx.x;
    const int lane = tid & 31;
    const int wid  = tid >> 5;
    const int wm   = (wid & 3) * 32;
    const int wn   = (wid >> 2) * 32;
    const int kgrp = lane >> 4;
    const int lrow = lane & 15;

    const int*   tIdx = tok_idx + e * T_TOK;
    const float* tW   = tok_w + e * T_TOK;

    // A rows are compacted+padded -> contiguous 2D tile at hb[baseS + mt*128].
    const unsigned short* aTile = hb + (size_t)(baseS + mt * 128) * F_DIM;
#if !HAVE_TDM
    const int segA = tid & 3;
    int rowA[2];
    const unsigned short* aSrc[2];
#pragma unroll
    for (int j = 0; j < 2; ++j) {
        rowA[j] = j * 64 + (tid >> 2);
        aSrc[j] = aTile + (size_t)rowA[j] * F_DIM + segA * 8;
    }
#endif
    const int segB = tid & 7;
    const float* bSrc[2];
#pragma unroll
    for (int j = 0; j < 2; ++j) {
        int row = j * 32 + (tid >> 3);
        bSrc[j] = w2 + (size_t)e * H_DIM * F_DIM
                     + (size_t)(nt * 64 + row) * F_DIM + segB * 4;
    }

    f32x8 acc[2][2];
#pragma unroll
    for (int i = 0; i < 2; ++i)
#pragma unroll
        for (int j = 0; j < 2; ++j) acc[i][j] = zero8();

    auto load_tile = [&](int buf, int k0) {
#if HAVE_TDM
        if (wid == 0)
            tdm_load_2d(lds_addr_of(&lA[buf][0]), aTile + k0,
                        /*tile_w=*/32, /*tile_h=*/128, /*stride=*/F_DIM);
#else
#pragma unroll
        for (int j = 0; j < 2; ++j) {
            u16x8 val = *(const u16x8*)(aSrc[j] + k0);
            *(u16x8*)(&lA[buf][rowA[j] * 32 + segA * 8]) = val;
        }
#endif
#pragma unroll
        for (int j = 0; j < 2; ++j) {
            int row = j * 32 + (tid >> 3);
            __builtin_prefetch(bSrc[j] + k0 + 96, 0, 1);
            float4 v = *(const float4*)(bSrc[j] + k0);
            u16x4 bv = {f2bf(v.x), f2bf(v.y), f2bf(v.z), f2bf(v.w)};
            *(u16x4*)(&lB[buf][row * 32 + segB * 4]) = bv;
        }
    };

    auto tensor_fence = [&]() {
#if HAVE_TDM
        if (wid == 0) __builtin_amdgcn_s_wait_tensorcnt(0);
#endif
    };

    load_tile(0, 0);
    tensor_fence();
    int cur = 0;
    for (int k0 = 0; k0 < F_DIM; k0 += 32) {
        __syncthreads();                       // tile `cur` ready (DS + TDM)
        if (k0 + 32 < F_DIM) load_tile(cur ^ 1, k0 + 32);

        bf16x16 a0 = load_frag(&lA[cur][(wm + 0 * 16 + lrow) * 32], kgrp);
        bf16x16 a1 = load_frag(&lA[cur][(wm + 1 * 16 + lrow) * 32], kgrp);
        bf16x16 b0 = load_frag(&lB[cur][(wn + 0 * 16 + lrow) * 32], kgrp);
        bf16x16 b1 = load_frag(&lB[cur][(wn + 1 * 16 + lrow) * 32], kgrp);

        acc[0][0] = wmma_bf16(a0, b0, acc[0][0]);
        acc[0][1] = wmma_bf16(a0, b1, acc[0][1]);
        acc[1][0] = wmma_bf16(a1, b0, acc[1][0]);
        acc[1][1] = wmma_bf16(a1, b1, acc[1][1]);

        tensor_fence();                        // TDM for next tile complete
        cur ^= 1;
    }

#pragma unroll
    for (int mi = 0; mi < 2; ++mi)
#pragma unroll
        for (int ni = 0; ni < 2; ++ni)
#pragma unroll
            for (int v = 0; v < 8; ++v) {
                int rowt = wm + mi * 16 + v + kgrp * 8;
                int p = mt * 128 + rowt;
                if (p < cnt) {
                    int t = tIdx[p];
                    float w = tW[p];
                    int col = nt * 64 + wn + ni * 16 + lrow;
                    atomicAdd(&out[(size_t)t * H_DIM + col], w * acc[mi][ni][v]);
                }
            }
}

// ---------------------------------------------------------------------------
// ws layout (bytes):
//   cnt    @ 0        (8 x i32)
//   base   @ 256      (8 x i32)
//   tokIdx @ 512      (E*T x i32,  64 KB)
//   tokW   @ 66048    (E*T x f32,  64 KB)
//   x_bf16 @ 131584   (T*H  x u16, 8 MB)
//   h_bf16 @ 8520192  ((2T+128) x F x u16, ~11.9 MB; +128 slots of tile pad)
// ---------------------------------------------------------------------------
extern "C" void kernel_launch(void* const* d_in, const int* in_sizes, int n_in,
                              void* d_out, int out_size, void* d_ws, size_t ws_size,
                              hipStream_t stream) {
    const float* x  = (const float*)d_in[0];
    const float* gw = (const float*)d_in[1];
    const float* w1 = (const float*)d_in[2];
    const float* w3 = (const float*)d_in[3];
    const float* w2 = (const float*)d_in[4];
    float* out = (float*)d_out;

    char* ws = (char*)d_ws;
    int*   cnt    = (int*)(ws + 0);
    int*   base   = (int*)(ws + 256);
    int*   tokIdx = (int*)(ws + 512);
    float* tokW   = (float*)(ws + 512 + 65536);
    unsigned short* xb = (unsigned short*)(ws + 131584);
    unsigned short* hb = xb + (size_t)T_TOK * H_DIM;

    float* logits_out = out + (size_t)T_TOK * H_DIM;

    zero_init<<<(T_TOK * H_DIM / 4 + 255) / 256, 256, 0, stream>>>(
        out, T_TOK * H_DIM / 4, cnt);
    router_kernel<<<T_TOK / 8, 256, 0, stream>>>(x, gw, logits_out, cnt, tokIdx, tokW);
    convert_x<<<T_TOK * H_DIM / 1024, 256, 0, stream>>>(x, xb);
    scan_base<<<1, 32, 0, stream>>>(cnt, base);
    moe_gemm1<<<dim3(F_DIM / 64, T_TOK / 128, E_NUM), 256, 0, stream>>>(
        xb, tokIdx, cnt, base, w1, w3, hb);
    moe_gemm2<<<dim3(H_DIM / 64, T_TOK / 128, E_NUM), 256, 0, stream>>>(
        hb, tokIdx, tokW, cnt, base, w2, out);
}
